// SCalculator_57183194579313
// MI455X (gfx1250) — matrix operand order repaired and checked by reading, not compiled, not run on hardware
//
#include <hip/hip_runtime.h>

// Problem constants (from the reference)
#define NTOT   100000
#define HDIM   150          // S == H == 150
#define KSEG   150          // K per segment; 4 segments concatenated
#define KC     32           // K chunk per WMMA step (bf16 16x16x32)
#define CPSEG  5            // ceil(150/32) chunks per segment
#define NCH    20           // 4 segments * 5 chunks
#define M_TILE 128          // rows of N per block
#define LSTR   40           // LDS A row stride in halfwords (80B: 16B aligned, bank-friendly)
#define ASZ    (M_TILE * LSTR)       // A buffer elements (bf16)

// Pre-converted weight layout in workspace: [gate 4][n 0..191][seg 4][k 0..159] bf16
// -> per-chunk fragment offset = (seg*KPAD + k0)*2 bytes <= 1216 B (fits load imm)
#define NPAD   192
#define KPAD   160
#define WS_NEEDED (4ull * 4 * NPAD * KPAD * 2)   // 983040 bytes

typedef __attribute__((ext_vector_type(16))) __bf16        v16bf;
typedef __attribute__((ext_vector_type(2)))  __bf16        v2bf;
typedef __attribute__((ext_vector_type(8)))  float         v8f;
typedef __attribute__((ext_vector_type(4)))  unsigned int  v4u;

union Frag16 {
    v4u   q[2];
    v16bf v;
};

__device__ __forceinline__ float sigmoidf(float x) {
    return 1.0f / (1.0f + __expf(-x));
}
__device__ __forceinline__ float tanh_fast(float x) {
    float e = __expf(2.0f * x);
    return 1.0f - 2.0f / (e + 1.0f);   // saturates cleanly at +/-1
}

// ---------------------------------------------------------------------------
// Phase 0: convert fp32 weights -> bf16 in WMMA-B-fragment order [g][n][seg][k]
// ---------------------------------------------------------------------------
__global__ __launch_bounds__(256) void convert_weights(
    const float* __restrict__ w_in, const float* __restrict__ w_out,
    const float* __restrict__ u_in, const float* __restrict__ u_out,
    __bf16* __restrict__ Wb)
{
    const float* wgts[4] = {w_in, w_out, u_in, u_out};
    const int o = blockIdx.x * 256 + threadIdx.x;       // writes coalesced along k
    if (o >= 4 * 4 * NPAD * KPAD) return;
    const int kk  = o % KPAD;
    const int seg = (o / KPAD) % 4;
    const int n   = (o / (KPAD * 4)) % NPAD;
    const int g   = o / (KPAD * 4 * NPAD);
    float v = 0.0f;
    if (kk < KSEG && n < HDIM)
        v = wgts[seg][(g * KSEG + kk) * HDIM + n];
    Wb[o] = (__bf16)v;
}

// ---------------------------------------------------------------------------
// Main GEMM path: all per-chunk variation is a SMALL compile-time immediate;
// 8 B-fragment base pointers + 2 A bases are computed exactly once.
// ---------------------------------------------------------------------------

// rb2[p]: per-thread byte offset (row*600 + (t&15)*8), chunk-invariant.
__device__ __forceinline__ void stage_A2_full(
    const float* __restrict__ act, const int rb2[8], int k0,
    __bf16* __restrict__ A_store)
{
    const char* base = (const char*)act + k0 * 4;       // scalar add (SADDR)
    #pragma unroll
    for (int p = 0; p < 8; ++p) {
        float2 v = *(const float2*)(base + rb2[p]);     // saddr + voffset
        v2bf pr = { (__bf16)v.x, (__bf16)v.y };
        *(v2bf*)(A_store + p * 16 * LSTR) = pr;         // constant DS offsets
    }
}

__device__ __forceinline__ void stage_A2_tail(
    const float* __restrict__ act, const int rb2[8], int k0, int t,
    __bf16* __restrict__ A_store)
{
    const int kk  = (t & 15) * 2;
    const int kg  = k0 + kk;
    const bool kok = (kg + 1) < KSEG;
    const int koff = kok ? (k0 * 4) : (-(kk * 4));      // clamp to row start
    #pragma unroll
    for (int p = 0; p < 8; ++p) {
        float2 v = *(const float2*)((const char*)act + rb2[p] + koff);
        if (!kok) { v.x = 0.0f; v.y = 0.0f; }           // cndmask, no branch
        v2bf pr = { (__bf16)v.x, (__bf16)v.y };
        *(v2bf*)(A_store + p * 16 * LSTR) = pr;
    }
}

__device__ __forceinline__ void compute_chunk2(
    const __bf16* const aa[2], int abuf,                // A frag bases + buffer offset
    const __bf16* const bb[4][2], int bofs,             // B frag bases + small (seg,k0) offset
    v8f acc[4][2][2])
{
    Frag16 a[2];
    #pragma unroll
    for (int mi = 0; mi < 2; ++mi) {
        a[mi].q[0] = *(const v4u*)(aa[mi] + abuf);       // K = lh*8 .. +7
        a[mi].q[1] = *(const v4u*)(aa[mi] + abuf + 16);  // K = 16+lh*8 .. +7
    }
    #pragma unroll
    for (int g = 0; g < 4; ++g) {
        Frag16 b[2];
        #pragma unroll
        for (int ni = 0; ni < 2; ++ni) {
            b[ni].q[0] = *(const v4u*)(bb[g][ni] + bofs);       // K = lh*8 .. +7
            b[ni].q[1] = *(const v4u*)(bb[g][ni] + bofs + 16);  // K = 16+lh*8 .. +7
        }
        #pragma unroll
        for (int mi = 0; mi < 2; ++mi)
            #pragma unroll
            for (int ni = 0; ni < 2; ++ni)
                acc[g][mi][ni] = __builtin_amdgcn_wmma_f32_16x16x32_bf16(
                    false, a[mi].v, false, b[ni].v,
                    (short)0, acc[g][mi][ni], false, false);
    }
}

__global__ __launch_bounds__(256) void lstm_fused_wmma2(
    const float* __restrict__ s_in,  const float* __restrict__ s_out,
    const float* __restrict__ h_in,  const float* __restrict__ h_out,
    const float* __restrict__ last_c,
    const __bf16* __restrict__ Wb,
    const float* __restrict__ bias,
    float* __restrict__ out)
{
    __shared__ __bf16 A_lds[2 * ASZ];       // 2 x 10 KB, double buffered

    const int t    = threadIdx.x;
    const int lane = t & 31;
    const int wave = t >> 5;
    const int wn   = wave & 1;              // 0..1 : 32-column half
    const int wm   = wave >> 1;             // 0..3 : 32-row quarter
    const int h0   = blockIdx.x * 64;       // 3 col tiles (fastest -> A L2 reuse)
    const int m0   = blockIdx.y * M_TILE;
    const int lm   = lane & 15;
    const int lh   = lane >> 4;

    const float* acts[4] = {s_in, s_out, h_in, h_out};

    // ---- one-time base addresses -------------------------------------------
    const __bf16* aa[2];
    #pragma unroll
    for (int mi = 0; mi < 2; ++mi)
        aa[mi] = A_lds + (wm * 32 + mi * 16 + lm) * LSTR + lh * 8;

    const __bf16* bb[4][2];
    #pragma unroll
    for (int g = 0; g < 4; ++g)
        #pragma unroll
        for (int ni = 0; ni < 2; ++ni) {
            const int n = h0 + wn * 32 + ni * 16 + lm;
            bb[g][ni] = Wb + ((g * NPAD + n) * 4) * KPAD + lh * 8;    // [g][n][seg][k]
        }

    int rb2[8];
    #pragma unroll
    for (int p = 0; p < 8; ++p) {
        int grow = m0 + p * 16 + (t >> 4);
        grow = (grow < NTOT) ? grow : (NTOT - 1);       // clamp; masked at store
        rb2[p] = grow * (KSEG * 4) + (t & 15) * 8;      // bytes
    }
    __bf16* sA = A_lds + (t >> 4) * LSTR + (t & 15) * 2;

    v8f acc[4][2][2] = {};                  // [gate][mi][ni]

    // ---- pipelined K loop: everything per-chunk is a small immediate -------
    stage_A2_full(acts[0], rb2, 0, sA);
    #pragma unroll
    for (int c = 0; c < NCH; ++c) {
        __syncthreads();
        if (c + 1 < NCH) {
            const int cn  = c + 1;
            const int seg = cn / CPSEG;
            const int k0  = (cn % CPSEG) * KC;
            __bf16* dst = sA + (cn & 1) * ASZ;
            if ((cn % CPSEG) == CPSEG - 1)
                stage_A2_tail(acts[seg], rb2, k0, t, dst);
            else
                stage_A2_full(acts[seg], rb2, k0, dst);
        }
        compute_chunk2(aa, (c & 1) * ASZ,
                       bb, (c / CPSEG) * KPAD + (c % CPSEG) * KC,   // <= 608 elements
                       acc);
    }

    // ---- fused epilogue: sigmoid gates, cell/hidden, streaming I/O ---------
    #pragma unroll
    for (int ni = 0; ni < 2; ++ni) {
        const int col = h0 + wn * 32 + ni * 16 + lm;
        if (col < HDIM) {
            float bg[4];
            #pragma unroll
            for (int g = 0; g < 4; ++g) bg[g] = bias[g * HDIM + col];
            #pragma unroll
            for (int mi = 0; mi < 2; ++mi) {
                #pragma unroll
                for (int i = 0; i < 8; ++i) {
                    const int row = m0 + wm * 32 + mi * 16 + lh * 8 + i;
                    if (row < NTOT) {
                        const float gi = sigmoidf(acc[0][mi][ni][i] + bg[0]);
                        const float go = sigmoidf(acc[1][mi][ni][i] + bg[1]);
                        const float gf = sigmoidf(acc[2][mi][ni][i] + bg[2]);
                        const float gu = sigmoidf(acc[3][mi][ni][i] + bg[3]);
                        const float lc = __builtin_nontemporal_load(&last_c[row * HDIM + col]);
                        const float cc = gf * lc + gu * gi;
                        const float hv = go * tanh_fast(cc);
                        __builtin_nontemporal_store(hv, &out[row * HDIM + col]);
                        __builtin_nontemporal_store(cc, &out[NTOT * HDIM + row * HDIM + col]);
                    }
                }
            }
        }
    }
}

// ---------------------------------------------------------------------------
// Fallback (self-contained, no workspace): CTILE=32, B transposed through LDS
// ---------------------------------------------------------------------------
#define CTILE_FB 32
#define BSZ_FB   (4 * CTILE_FB * LSTR)

__device__ __forceinline__ void stage_chunk_fb(
    const float* __restrict__ act, const float* __restrict__ wgt,
    int m0, int h0, int k0, int t,
    __bf16* __restrict__ A_buf, __bf16* __restrict__ B_buf)
{
    {
        const int kk  = (t & 15) * 2;
        const int kg  = k0 + kk;
        const bool kok = (kg + 1) < KSEG;
        const int kgs = kok ? kg : 0;
        const int rb  = t >> 4;
        #pragma unroll
        for (int p = 0; p < 8; ++p) {
            const int r = p * 16 + rb;
            int grow = m0 + r;
            grow = (grow < NTOT) ? grow : (NTOT - 1);
            float2 v = *(const float2*)(act + grow * KSEG + kgs);
            if (!kok) { v.x = 0.0f; v.y = 0.0f; }
            v2bf pr = { (__bf16)v.x, (__bf16)v.y };
            *(v2bf*)(A_buf + r * LSTR + kk) = pr;
        }
    }
    {
        const int nb  = (t & 15) * 2;
        const int h   = h0 + nb;
        const bool hok = (h + 1) < HDIM;
        const int hs  = hok ? h : 0;
        const int rem = t >> 4;
        #pragma unroll
        for (int p = 0; p < 8; ++p) {
            const int idx = p * 16 + rem;
            const int g   = idx >> 5;
            const int kk  = idx & 31;
            const int kg  = k0 + kk;
            const bool kok = kg < KSEG;
            const int kgs = kok ? kg : 0;
            float2 w = *(const float2*)(wgt + (g * KSEG + kgs) * HDIM + hs);
            if (!(hok && kok)) { w.x = 0.0f; w.y = 0.0f; }
            B_buf[(g * CTILE_FB + nb    ) * LSTR + kk] = (__bf16)w.x;
            B_buf[(g * CTILE_FB + nb + 1) * LSTR + kk] = (__bf16)w.y;
        }
    }
}

__device__ __forceinline__ void compute_chunk_fb(
    const __bf16* __restrict__ A_buf, const __bf16* __restrict__ B_buf,
    int wm, int wn, int lane, v8f acc[4][2])
{
    const int lm = lane & 15;
    const int lh = lane >> 4;
    Frag16 a[2], bf[4];
    #pragma unroll
    for (int mi = 0; mi < 2; ++mi) {
        const int m = wm * 32 + mi * 16 + lm;
        const __bf16* base = A_buf + m * LSTR + lh * 8;
        a[mi].q[0] = *(const v4u*)(base);
        a[mi].q[1] = *(const v4u*)(base + 16);
    }
    #pragma unroll
    for (int g = 0; g < 4; ++g) {
        const int n = wn * 16 + lm;
        const __bf16* base = B_buf + (g * CTILE_FB + n) * LSTR + lh * 8;
        bf[g].q[0] = *(const v4u*)(base);
        bf[g].q[1] = *(const v4u*)(base + 16);
    }
    #pragma unroll
    for (int g = 0; g < 4; ++g)
        #pragma unroll
        for (int mi = 0; mi < 2; ++mi)
            acc[g][mi] = __builtin_amdgcn_wmma_f32_16x16x32_bf16(
                false, a[mi].v, false, bf[g].v,
                (short)0, acc[g][mi], false, false);
}

__global__ __launch_bounds__(256) void lstm_fused_wmma_fb(
    const float* __restrict__ s_in,  const float* __restrict__ s_out,
    const float* __restrict__ h_in,  const float* __restrict__ h_out,
    const float* __restrict__ last_c,
    const float* __restrict__ w_in,  const float* __restrict__ w_out,
    const float* __restrict__ u_in,  const float* __restrict__ u_out,
    const float* __restrict__ bias,
    float* __restrict__ out)
{
    __shared__ __bf16 A_lds[2 * ASZ];
    __shared__ __bf16 B_lds[2 * BSZ_FB];

    const int t    = threadIdx.x;
    const int lane = t & 31;
    const int wave = t >> 5;
    const int wn   = wave & 1;
    const int wm   = wave >> 1;
    const int h0   = blockIdx.x * CTILE_FB;
    const int m0   = blockIdx.y * M_TILE;

    const float* acts[4] = {s_in, s_out, h_in, h_out};
    const float* wgts[4] = {w_in, w_out, u_in, u_out};

    v8f acc[4][2] = {};

    stage_chunk_fb(acts[0], wgts[0], m0, h0, 0, t, A_lds, B_lds);
    #pragma unroll
    for (int c = 0; c < NCH; ++c) {
        __syncthreads();
        if (c + 1 < NCH) {
            const int seg = (c + 1) / CPSEG;
            const int k0  = ((c + 1) % CPSEG) * KC;
            stage_chunk_fb(acts[seg], wgts[seg], m0, h0, k0, t,
                           A_lds + ((c + 1) & 1) * ASZ, B_lds + ((c + 1) & 1) * BSZ_FB);
        }
        compute_chunk_fb(A_lds + (c & 1) * ASZ, B_lds + (c & 1) * BSZ_FB,
                         wm, wn, lane, acc);
    }

    const int lm  = lane & 15;
    const int lh  = lane >> 4;
    const int col = h0 + wn * 16 + lm;
    if (col < HDIM) {
        float bg[4];
        #pragma unroll
        for (int g = 0; g < 4; ++g) bg[g] = bias[g * HDIM + col];
        #pragma unroll
        for (int mi = 0; mi < 2; ++mi) {
            #pragma unroll
            for (int i = 0; i < 8; ++i) {
                const int row = m0 + wm * 32 + mi * 16 + lh * 8 + i;
                if (row < NTOT) {
                    const float gi = sigmoidf(acc[0][mi][i] + bg[0]);
                    const float go = sigmoidf(acc[1][mi][i] + bg[1]);
                    const float gf = sigmoidf(acc[2][mi][i] + bg[2]);
                    const float gu = sigmoidf(acc[3][mi][i] + bg[3]);
                    const float cc = gf * last_c[row * HDIM + col] + gu * gi;
                    const float hv = go * tanh_fast(cc);
                    out[row * HDIM + col]               = hv;
                    out[NTOT * HDIM + row * HDIM + col] = cc;
                }
            }
        }
    }
}

extern "C" void kernel_launch(void* const* d_in, const int* in_sizes, int n_in,
                              void* d_out, int out_size, void* d_ws, size_t ws_size,
                              hipStream_t stream) {
    (void)in_sizes; (void)n_in; (void)out_size;
    const float* s_in   = (const float*)d_in[0];
    const float* s_out  = (const float*)d_in[1];
    const float* h_in   = (const float*)d_in[2];
    const float* h_out  = (const float*)d_in[3];
    const float* last_c = (const float*)d_in[4];
    const float* w_in   = (const float*)d_in[5];
    const float* w_out  = (const float*)d_in[6];
    const float* u_in   = (const float*)d_in[7];
    const float* u_out  = (const float*)d_in[8];
    const float* bias   = (const float*)d_in[9];
    float* out = (float*)d_out;

    const int row_tiles = (NTOT + M_TILE - 1) / M_TILE;     // 782

    if (ws_size >= WS_NEEDED && d_ws != nullptr) {
        __bf16* Wb = (__bf16*)d_ws;
        const int nconv = 4 * 4 * NPAD * KPAD;
        convert_weights<<<dim3((nconv + 255) / 256), dim3(256), 0, stream>>>(
            w_in, w_out, u_in, u_out, Wb);
        lstm_fused_wmma2<<<dim3(3, row_tiles), dim3(256), 0, stream>>>(
            s_in, s_out, h_in, h_out, last_c, (const __bf16*)Wb, bias, out);
    } else {
        lstm_fused_wmma_fb<<<dim3(5, row_tiles), dim3(256), 0, stream>>>(
            s_in, s_out, h_in, h_out, last_c, w_in, w_out, u_in, u_out, bias, out);
    }
}
